// HGCNN_norm_geo_3513283248952
// MI455X (gfx1250) — compile-verified, matrix operand order checked by reading, not run on hardware
//
#include <hip/hip_runtime.h>
#include <hip/hip_fp16.h>

typedef __attribute__((ext_vector_type(2)))  float    v2f;
typedef __attribute__((ext_vector_type(8)))  float    v8f;
typedef __attribute__((ext_vector_type(16))) _Float16 v16h;

#define BN_SCALE 0.99999500003749977f   /* 1/sqrt(1+1e-5) */

constexpr int NPTS = 2048;   // points per batch (problem constant)
constexpr int KNB  = 20;     // neighbors (problem constant)

// K-index mapping inside a 32-wide chunk for 16-bit WMMA operands (ISA 7.12.2):
// VGPR v (0..7), lane-half h, halfword hl ->  K = 16*(v/4) + 8*h + 2*(v%4) + hl
__device__ __forceinline__ int kmap32(int v, int half, int hl) {
  return ((v >> 2) << 4) + (half << 3) + ((v & 3) << 1) + hl;
}

// ---------------------------------------------------------------------------
// xx[b,n] = sum_c x[b,c,n]^2
// ---------------------------------------------------------------------------
__global__ void rownorm_kernel(const float* __restrict__ xin, float* __restrict__ xx,
                               int choff, int incb, int Cin) {
  int b = blockIdx.y;
  int n = blockIdx.x * blockDim.x + threadIdx.x;
  const float* p = xin + ((size_t)b * incb + choff) * NPTS + n;
  float s = 0.f;
  for (int c = 0; c < Cin; ++c) {
    float v = p[c * NPTS];
    s += v * v;
  }
  xx[b * NPTS + n] = s;
}

// ---------------------------------------------------------------------------
// Fused kNN: 16-row tile of pd = 2*X^T X - xx_i - xx_j via f32 WMMA, kept in
// LDS (16 x 2048 f32 = 128KB), then row-min, mask substitution, iterative
// top-k (lex order matches jax.lax.top_k ties).
// Grid: (NPTS/16, B), 256 threads (8 waves). Templated on CIN so all chunk
// loops fully unroll: A fragments stay in VGPRs, bounds checks constant-fold.
// ---------------------------------------------------------------------------
template <int CIN>
__global__ __launch_bounds__(256, 1) void knn_topk_kernel(
    const float* __restrict__ xin, const float* __restrict__ xx,
    const unsigned char* __restrict__ mask, int* __restrict__ idx_out,
    int choff, int incb, int localFlag) {
  __shared__ float pd[16 * NPTS];
  __shared__ float mn[16];

  constexpr int NCH = (CIN + 3) >> 2;   // K chunks of 4 for f32 WMMA

  const int b = blockIdx.y;
  const int rowBase = blockIdx.x * 16;
  const int tid = threadIdx.x;
  const int lane = tid & 31;
  const int wave = tid >> 5;
  const int lmod = lane & 15;
  const int lhalf = lane >> 4;
  const float* xb = xin + ((size_t)b * incb + choff) * NPTS;

  // ---- Phase 1: Gram matrix tile via V_WMMA_F32_16X16X4_F32 ----
  v2f aF[NCH];
#pragma unroll
  for (int c = 0; c < NCH; ++c) {
#pragma unroll
    for (int v = 0; v < 2; ++v) {
      int K = c * 4 + lhalf * 2 + v;
      int Kc = (K < CIN) ? K : 0;                 // clamp, no branch
      float val = xb[Kc * NPTS + rowBase + lmod];
      aF[c][v] = (K < CIN) ? val : 0.f;
    }
  }
  float xxrow[8];
#pragma unroll
  for (int v = 0; v < 8; ++v) xxrow[v] = xx[b * NPTS + rowBase + v + 8 * lhalf];

  for (int ct = wave; ct < (NPTS >> 4); ct += 8) {
    const int col = ct * 16 + lmod;
    if (ct + 8 < (NPTS >> 4))
      __builtin_prefetch(xb + (ct + 8) * 16 + lmod, 0, 1);
    v8f acc = {};
#if __has_builtin(__builtin_amdgcn_wmma_f32_16x16x4_f32)
#pragma unroll
    for (int c = 0; c < NCH; ++c) {
      v2f bF;
#pragma unroll
      for (int v = 0; v < 2; ++v) {
        int K = c * 4 + lhalf * 2 + v;
        int Kc = (K < CIN) ? K : 0;
        float val = xb[Kc * NPTS + col];
        bF[v] = (K < CIN) ? val : 0.f;
      }
      acc = __builtin_amdgcn_wmma_f32_16x16x4_f32(false, aF[c], false, bF,
                                                  (short)0, acc, false, false);
    }
#else
#pragma unroll
    for (int v = 0; v < 8; ++v) {
      int row = rowBase + v + 8 * lhalf;
      float s = 0.f;
      for (int K = 0; K < CIN; ++K)
        s += xb[K * NPTS + row] * xb[K * NPTS + col];
      acc[v] = s;
    }
#endif
    const float xxc = xx[b * NPTS + col];
#pragma unroll
    for (int v = 0; v < 8; ++v) {
      int row = v + 8 * lhalf;
      pd[row * NPTS + col] = 2.f * acc[v] - xxrow[v] - xxc;
    }
  }
  __syncthreads();

  // ---- Phase 2: per-row min (BEFORE masking, as in reference) ----
  const int row = tid >> 4;   // 0..15
  const int ln = tid & 15;    // 16 lanes per row
  {
    float m = 3.4e38f;
    for (int i = 0; i < (NPTS >> 4); ++i)
      m = fminf(m, pd[row * NPTS + ln + (i << 4)]);
#pragma unroll
    for (int off = 8; off; off >>= 1) m = fminf(m, __shfl_xor(m, off, 16));
    if (ln == 0) mn[row] = m;
  }
  __syncthreads();

  // ---- Phase 3: mask substitution ----
  {
    const unsigned char* mk = mask + b * NPTS;
    for (int e = tid; e < 16 * NPTS; e += 256) {
      int r = e >> 11, c = e & (NPTS - 1);
      float v = pd[e];
      bool mm = mk[c] != 0;
      float repl = mn[r];
      v = localFlag ? (mm ? v : repl) : (mm ? repl : v);
      pd[e] = v;
    }
  }
  __syncthreads();

  // ---- Phase 4: iterative top-k, threshold form (no LDS mutation) ----
  {
    float prevV = 3.4e38f;
    int prevI = -1;
    int* orow = idx_out + ((size_t)(b * NPTS + rowBase + row)) * KNB;
    const float* prow = pd + row * NPTS;
    for (int sel = 0; sel < KNB; ++sel) {
      float best = -3.4e38f;
      int bi = NPTS;
      for (int i = 0; i < (NPTS >> 4); ++i) {
        int c = ln + (i << 4);
        float v = prow[c];
        bool elig = (v < prevV) || (v == prevV && c > prevI);
        if (elig && (v > best || (v == best && c < bi))) { best = v; bi = c; }
      }
#pragma unroll
      for (int off = 8; off; off >>= 1) {
        float ov = __shfl_xor(best, off, 16);
        int oi = __shfl_xor(bi, off, 16);
        if (ov > best || (ov == best && oi < bi)) { best = ov; bi = oi; }
      }
      if (ln == 0) orow[sel] = bi;
      prevV = best;
      prevI = bi;
    }
  }
}

// ---------------------------------------------------------------------------
// Fused edge conv: gather [feat-ctr, ctr] -> GEMM (f16 WMMA, f32 acc) ->
// BN + LReLU -> max over k. One wave: 4 points x 16 out channels
// (5 column tiles of 16; 4*20 = 80 cols). Weights loaded once per K-chunk,
// reused across the 5 B-tiles. Templated on CIN: chunk loop fully unrolls,
// all channel-range predicates constant-fold (CIN=64/128) or become cndmask.
// Grid: (NPTS/4, O/16, B), 32 threads.
// ---------------------------------------------------------------------------
template <int CIN>
__global__ __launch_bounds__(32, 1) void edge_conv_kernel(
    const float* __restrict__ xin, const int* __restrict__ nidx,
    const float* __restrict__ W, const float* __restrict__ gg,
    const float* __restrict__ bb, float* __restrict__ out,
    int inoff, int incb, int outoff, int outcb) {
  __shared__ float ybuf[16][80];
  constexpr int K2 = 2 * CIN;
  constexpr int NCH = (K2 + 31) >> 5;

  const int b = blockIdx.z;
  const int otile = blockIdx.y;
  const int nbase = blockIdx.x * 4;
  const int lane = threadIdx.x;
  const int lmod = lane & 15, lhalf = lane >> 4;
  const float* xb = xin + ((size_t)b * incb + inoff) * NPTS;
  const int O = gridDim.y * 16;

  const float* pnT[5];   // center point column base
  const float* pbT[5];   // neighbor point column base
#pragma unroll
  for (int t = 0; t < 5; ++t) {
    int j = t * 16 + lmod;          // local col 0..79
    int n = nbase + j / KNB;
    int kk = j % KNB;
    pnT[t] = xb + n;
    pbT[t] = xb + nidx[(b * NPTS + n) * KNB + kk];
  }
  const int orow = otile * 16 + lmod;

  v8f acc[5] = {};
#pragma unroll
  for (int c0 = 0; c0 < NCH; ++c0) {
    v16h aF;
#pragma unroll
    for (int v = 0; v < 8; ++v)
#pragma unroll
      for (int hl = 0; hl < 2; ++hl) {
        int cc = (c0 << 5) + kmap32(v, lhalf, hl);
        int wa = (cc < K2) ? cc : 0;
        float wv = W[orow * K2 + wa];
        aF[2 * v + hl] = (_Float16)((cc < K2) ? wv : 0.f);
      }
#pragma unroll
    for (int t = 0; t < 5; ++t) {
      v16h bF;
#pragma unroll
      for (int v = 0; v < 8; ++v)
#pragma unroll
        for (int hl = 0; hl < 2; ++hl) {
          int cc = (c0 << 5) + kmap32(v, lhalf, hl);
          int cd = (cc < CIN) ? cc : 0;                       // diff channel
          int ce = (cc >= CIN && cc < K2) ? (cc - CIN) : 0;   // center channel
          float diff = pbT[t][cd * NPTS] - pnT[t][cd * NPTS];
          float ctr = pnT[t][ce * NPTS];
          float fv = (cc < CIN) ? diff : ((cc < K2) ? ctr : 0.f);
          bF[2 * v + hl] = (_Float16)fv;
        }
      acc[t] = __builtin_amdgcn_wmma_f32_16x16x32_f16(false, aF, false, bF,
                                                      (short)0, acc[t],
                                                      false, false);
    }
  }

  // BN + LReLU, stage into LDS
#pragma unroll
  for (int t = 0; t < 5; ++t)
#pragma unroll
    for (int v = 0; v < 8; ++v) {
      int ol = v + 8 * lhalf;
      int o = otile * 16 + ol;
      float y = acc[t][v] * (gg[o] * BN_SCALE) + bb[o];
      y = (y >= 0.f) ? y : 0.2f * y;
      ybuf[ol][t * 16 + lmod] = y;
    }
  __syncthreads();

  // max over k, single store (no atomics, no output init)
  for (int p = lane; p < 64; p += 32) {
    int ol = p >> 2, nl = p & 3;
    float mx = -3.4e38f;
#pragma unroll
    for (int kk = 0; kk < KNB; ++kk) mx = fmaxf(mx, ybuf[ol][nl * KNB + kk]);
    out[((size_t)b * outcb + outoff + otile * 16 + ol) * NPTS + nbase + nl] = mx;
  }
}

// ---------------------------------------------------------------------------
// W9 1x1 conv over masked-combined branches: in[c,n] = mask[n]?xl:xg,
// y[b,o,n] = lrelu(W9 @ in * g9s + b9).  Grid: (NPTS/16, 1024/16, B), 32 thr.
// ---------------------------------------------------------------------------
__global__ __launch_bounds__(32, 1) void conv9_kernel(
    const float* __restrict__ xl, const float* __restrict__ xg,
    const unsigned char* __restrict__ mask, const float* __restrict__ W,
    const float* __restrict__ gg, const float* __restrict__ bb,
    float* __restrict__ y) {
  const int b = blockIdx.z, otile = blockIdx.y, ctile = blockIdx.x;
  const int lane = threadIdx.x, lmod = lane & 15, lhalf = lane >> 4;
  const int col = ctile * 16 + lmod;
  const float* src = (mask[b * NPTS + col] != 0) ? xl : xg;
  src += (size_t)b * 512 * NPTS + col;
  const int orow = otile * 16 + lmod;

  v8f acc = {};
#pragma unroll
  for (int c0 = 0; c0 < 16; ++c0) {     // K = 512 = 16 chunks of 32
    v16h aF, bF;
#pragma unroll
    for (int v = 0; v < 8; ++v)
#pragma unroll
      for (int hl = 0; hl < 2; ++hl) {
        int cc = (c0 << 5) + kmap32(v, lhalf, hl);
        aF[2 * v + hl] = (_Float16)W[orow * 512 + cc];
        bF[2 * v + hl] = (_Float16)src[cc * NPTS];
      }
    acc = __builtin_amdgcn_wmma_f32_16x16x32_f16(false, aF, false, bF,
                                                 (short)0, acc, false, false);
  }
#pragma unroll
  for (int v = 0; v < 8; ++v) {
    int o = otile * 16 + v + 8 * lhalf;
    float yy = acc[v] * (gg[o] * BN_SCALE) + bb[o];
    yy = (yy >= 0.f) ? yy : 0.2f * yy;
    y[((size_t)b * 1024 + o) * NPTS + col] = yy;
  }
}

// ---------------------------------------------------------------------------
// feats[b, o] = max_n y[b,o,n];  feats[b, 1024+o] = mean_n y[b,o,n]
// Grid: B*1024, 256 threads.
// ---------------------------------------------------------------------------
__global__ __launch_bounds__(256) void reduce_maxmean_kernel(
    const float* __restrict__ y, float* __restrict__ feats) {
  const int bo = blockIdx.x;
  const int tid = threadIdx.x;
  const float* p = y + (size_t)bo * NPTS;
  float mx = -3.4e38f, sm = 0.f;
  for (int i = tid; i < NPTS; i += 256) {
    float v = p[i];
    mx = fmaxf(mx, v);
    sm += v;
  }
  __shared__ float smx[256], ssm[256];
  smx[tid] = mx;
  ssm[tid] = sm;
  __syncthreads();
  for (int s2 = 128; s2; s2 >>= 1) {
    if (tid < s2) {
      smx[tid] = fmaxf(smx[tid], smx[tid + s2]);
      ssm[tid] += ssm[tid + s2];
    }
    __syncthreads();
  }
  if (tid == 0) {
    int b = bo >> 10, o = bo & 1023;
    feats[b * 2048 + o] = smx[0];
    feats[b * 2048 + 1024 + o] = ssm[0] / (float)NPTS;
  }
}

// ---------------------------------------------------------------------------
// Small FC: out = [lrelu]((in @ W^T [+ bias]) [*g*s + b])
// ---------------------------------------------------------------------------
__global__ void fc_kernel(const float* __restrict__ in, const float* __restrict__ W,
                          const float* __restrict__ bias, const float* __restrict__ gg,
                          const float* __restrict__ bb, float* __restrict__ out,
                          int B, int In, int Out, int doBnLrelu) {
  int t = blockIdx.x * blockDim.x + threadIdx.x;
  if (t >= B * Out) return;
  int b = t / Out, o = t % Out;
  const float* iv = in + (size_t)b * In;
  const float* wv = W + (size_t)o * In;
  float s = 0.f;
  for (int i = 0; i < In; ++i) s += iv[i] * wv[i];
  if (bias) s += bias[o];
  if (doBnLrelu) {
    s = s * (gg[o] * BN_SCALE) + bb[o];
    s = (s >= 0.f) ? s : 0.2f * s;
  }
  out[t] = s;
}

// ---------------------------------------------------------------------------
extern "C" void kernel_launch(void* const* d_in, const int* in_sizes, int n_in,
                              void* d_out, int out_size, void* d_ws, size_t ws_size,
                              hipStream_t stream) {
  (void)in_sizes; (void)out_size; (void)ws_size;
  const int B = 8, N = NPTS;

  // Input order: x, geod_dist, W1..W9, (g,b)x11, L1, L2, L2b, L3, L3b, local_idx
  const float* x = (const float*)d_in[0];
  const float* Wp[9];
  for (int i = 0; i < 9; ++i) Wp[i] = (const float*)d_in[2 + i];
  const float *gP[11], *bP[11];
  for (int i = 0; i < 11; ++i) {
    gP[i] = (const float*)d_in[11 + 2 * i];
    bP[i] = (const float*)d_in[12 + 2 * i];
  }
  const float* L1 = (const float*)d_in[33];
  const float* L2 = (const float*)d_in[34];
  const float* L2b = (const float*)d_in[35];
  const float* L3 = (const float*)d_in[36];
  const float* L3b = (const float*)d_in[37];
  const unsigned char* mask = (const unsigned char*)d_in[n_in - 1];
  float* out = (float*)d_out;

  // Workspace partition
  size_t off = 0;
  auto take = [&](size_t bytes) {
    void* p = (char*)d_ws + off;
    off = (off + bytes + 255) & ~(size_t)255;
    return p;
  };
  float* xx = (float*)take((size_t)B * N * 4);
  int* nidx = (int*)take((size_t)B * N * KNB * 4);
  float* xl = (float*)take((size_t)B * 512 * N * 4);
  float* xg = (float*)take((size_t)B * 512 * N * 4);
  float* ybig = (float*)take((size_t)B * 1024 * N * 4);
  float* feats = (float*)take((size_t)B * 2048 * 4);
  float* h1 = (float*)take((size_t)B * 512 * 4);
  float* h2 = (float*)take((size_t)B * 256 * 4);

  struct BlkCfg { int inoff, incb, Cin, outoff, O; };
  const BlkCfg cfg[4] = {
      {0, 6, 6, 0, 64},
      {0, 512, 64, 64, 64},
      {64, 512, 64, 128, 128},
      {128, 512, 128, 256, 256},
  };

  for (int branch = 0; branch < 2; ++branch) {
    float* bout = (branch == 0) ? xl : xg;
    int localFlag = (branch == 0) ? 1 : 0;
    for (int i = 0; i < 4; ++i) {
      const BlkCfg c = cfg[i];
      const float* src = (i == 0) ? x : bout;
      const float* Wc = Wp[branch * 4 + i];
      const float* gc = gP[branch * 4 + i];
      const float* bc = bP[branch * 4 + i];

      rownorm_kernel<<<dim3(N / 256, B), 256, 0, stream>>>(src, xx, c.inoff,
                                                           c.incb, c.Cin);
      dim3 gknn(N / 16, B);
      dim3 gec(N / 4, c.O / 16, B);
      switch (c.Cin) {
        case 6:
          knn_topk_kernel<6><<<gknn, 256, 0, stream>>>(src, xx, mask, nidx,
                                                       c.inoff, c.incb, localFlag);
          edge_conv_kernel<6><<<gec, 32, 0, stream>>>(src, nidx, Wc, gc, bc,
                                                      bout, c.inoff, c.incb,
                                                      c.outoff, 512);
          break;
        case 64:
          knn_topk_kernel<64><<<gknn, 256, 0, stream>>>(src, xx, mask, nidx,
                                                        c.inoff, c.incb, localFlag);
          edge_conv_kernel<64><<<gec, 32, 0, stream>>>(src, nidx, Wc, gc, bc,
                                                       bout, c.inoff, c.incb,
                                                       c.outoff, 512);
          break;
        default:
          knn_topk_kernel<128><<<gknn, 256, 0, stream>>>(src, xx, mask, nidx,
                                                         c.inoff, c.incb, localFlag);
          edge_conv_kernel<128><<<gec, 32, 0, stream>>>(src, nidx, Wc, gc, bc,
                                                        bout, c.inoff, c.incb,
                                                        c.outoff, 512);
          break;
      }
    }
  }

  conv9_kernel<<<dim3(N / 16, 1024 / 16, B), 32, 0, stream>>>(
      xl, xg, mask, Wp[8], gP[8], bP[8], ybig);
  reduce_maxmean_kernel<<<B * 1024, 256, 0, stream>>>(ybig, feats);

  fc_kernel<<<(B * 512 + 255) / 256, 256, 0, stream>>>(
      feats, L1, nullptr, gP[9], bP[9], h1, B, 2048, 512, 1);
  fc_kernel<<<(B * 256 + 255) / 256, 256, 0, stream>>>(
      h1, L2, L2b, gP[10], bP[10], h2, B, 512, 256, 1);
  fc_kernel<<<(B * 40 + 255) / 256, 256, 0, stream>>>(
      h2, L3, L3b, nullptr, nullptr, out, B, 256, 40, 0);
}
